// AMS_33036888441121
// MI455X (gfx1250) — matrix operand order repaired and checked by reading, not compiled, not run on hardware
//
#include <hip/hip_runtime.h>
#include <hip/hip_bf16.h>
#include <math.h>

#define Bsz 64
#define Lsz 192
#define Nsz 862
#define Esz 4
#define NCHUNK 27          // ceil(862/32)
#define LP 200             // LDS row pitch in f16 elems for GEMM tiles

typedef __attribute__((ext_vector_type(16))) _Float16 v16h;
typedef __attribute__((ext_vector_type(8)))  float    v8f;
typedef __attribute__((ext_vector_type(2)))  float    v2f;

union AV { v16h v; unsigned int u[8]; };

__device__ __forceinline__ int clampi(int v, int lo, int hi) {
  return v < lo ? lo : (v > hi ? hi : v);
}

// ---------------------------------------------------------------------------
// Kernel 1: per (b, n-chunk of 32):
//   spec(192x32) = T(192x192) * xs(192x32) via V_WMMA_F32_16X16X4_F32
//     rows 0..95  of T:  cos(2*pi*k*t/192)      -> Re X[k]
//     rows 96..191 of T: -sin(2*pi*(k-96)*t/192) -> Im X[k-96]
//   then per-series top-3 |X|^2, seasonal reconstruction, trend (3 moving
//   averages), and deterministic partial reduce of gx = sum_n new_x*w_start.
// ---------------------------------------------------------------------------
__global__ __launch_bounds__(256)
void k_decomp(const float* __restrict__ x, const float* __restrict__ w_start,
              float* __restrict__ gxpart) {
  const int chunk = blockIdx.x;
  const int b = blockIdx.y;
  const int n0 = chunk * 32;
  const int tid = threadIdx.x;

  __shared__ float xs[Lsz * 33];        // [t][j], pitch 33
  __shared__ float cosT[Lsz];
  __shared__ float sinT[Lsz];
  __shared__ float sinN[Lsz];           // -sin
  __shared__ float spec[Lsz * 32];      // rows 0..95 re, 96..191 im
  __shared__ float selRe[32 * 3];
  __shared__ float selIm[32 * 3];
  __shared__ int   selK [32 * 3];

  if (tid < Lsz) {
    float ang = 6.28318530717958647692f * (float)tid / (float)Lsz;
    float s, c;
    __sincosf(ang, &s, &c);
    cosT[tid] = c;
    sinT[tid] = s;
    sinN[tid] = -s;
  }
  for (int e = tid; e < Lsz * 32; e += 256) {
    int l = e >> 5, j = e & 31;
    int n = n0 + j;
    xs[l * 33 + j] = (n < Nsz) ? x[(size_t)(b * Lsz + l) * Nsz + n] : 0.0f;
  }
  __syncthreads();

  // ---- WMMA rDFT: 24 16x16 output tiles (12 m-tiles x 2 n-tiles), 8 waves,
  // ---- 3 m-tiles per wave, K=192 in 48 steps of 4.
  {
    const int lane = tid & 31;
    const int wv   = tid >> 5;
    const int half = lane >> 4;
    const int mr   = lane & 15;
    const int nl   = lane & 15;
    const int ntile  = wv & 1;
    const int mtBase = (wv >> 1) * 3;
    const int jcol   = ntile * 16 + nl;

    // Per-tile twiddle generators: A[row][t] = tw[(kr*t) mod 192]
    int kr_[3], idx_[3], inc_[3];
    const float* tw_[3];
    #pragma unroll
    for (int q = 0; q < 3; ++q) {
      const int mt  = mtBase + q;
      const int row = mt * 16 + mr;
      const bool isSin = (mt >= 6);              // tile boundary = row 96
      const int kr = isSin ? (row - 96) : row;
      kr_[q]  = kr;
      tw_[q]  = isSin ? sinN : cosT;
      idx_[q] = (kr * (half * 2)) % Lsz;         // t at kk=0 is half*2
      inc_[q] = (kr * 4) % Lsz;                  // t advances by 4 per k-step
    }

    v8f accA[3] = {};
    for (int kk = 0; kk < 48; ++kk) {
      const int t0 = kk * 4 + half * 2;          // B rows K = half*2 + {0,1}
      v2f bf;
      bf[0] = xs[t0 * 33 + jcol];
      bf[1] = xs[t0 * 33 + 33 + jcol];
      #pragma unroll
      for (int q = 0; q < 3; ++q) {
        int i0 = idx_[q];
        int i1 = i0 + kr_[q]; if (i1 >= Lsz) i1 -= Lsz;
        v2f af;
        af[0] = tw_[q][i0];
        af[1] = tw_[q][i1];
        accA[q] = __builtin_amdgcn_wmma_f32_16x16x4_f32(
            false, af, false, bf, (short)0, accA[q], false, false);
        i0 += inc_[q]; if (i0 >= Lsz) i0 -= Lsz;
        idx_[q] = i0;
      }
    }
    // D layout: VGPR r -> row = tile*16 + r + 8*half, col = jcol
    #pragma unroll
    for (int q = 0; q < 3; ++q) {
      const int rowBase = (mtBase + q) * 16 + 8 * half;
      #pragma unroll
      for (int r = 0; r < 8; ++r)
        spec[(rowBase + r) * 32 + jcol] = accA[q][r];
    }
  }
  __syncthreads();

  // Top-3 |X|^2 per series, k=1..95; strict '>' => lowest-index tie-break.
  if (tid < 32) {
    const int j = tid;
    float m1 = -1.0f, m2 = -1.0f, m3 = -1.0f;
    int k1 = 1, k2 = 1, k3 = 1;
    for (int k = 1; k <= 95; ++k) {
      float re = spec[k * 32 + j];
      float im = spec[(96 + k) * 32 + j];
      float m = re * re + im * im;
      if (m > m1)      { m3 = m2; k3 = k2; m2 = m1; k2 = k1; m1 = m; k1 = k; }
      else if (m > m2) { m3 = m2; k3 = k2; m2 = m;  k2 = k; }
      else if (m > m3) { m3 = m;  k3 = k; }
    }
    int ks[3] = {k1, k2, k3};
    for (int c = 0; c < 3; ++c) {
      selK [j * 3 + c] = ks[c];
      selRe[j * 3 + c] = spec[ks[c] * 32 + j];
      selIm[j * 3 + c] = spec[(96 + ks[c]) * 32 + j];
    }
  }
  __syncthreads();

  // trend + sea + weighted reduce over this n-chunk (deterministic).
  if (tid < Lsz) {
    const int t = tid;
    float acc = 0.0f;
    for (int j = 0; j < 32; ++j) {
      int n = n0 + j;
      if (n >= Nsz) break;
      float s4 = 0.0f, s8 = 0.0f, s12 = 0.0f;
      for (int d = 0; d < 4;  ++d) s4  += xs[clampi(t - 1 + d, 0, Lsz - 1) * 33 + j];
      for (int d = 0; d < 8;  ++d) s8  += xs[clampi(t - 3 + d, 0, Lsz - 1) * 33 + j];
      for (int d = 0; d < 12; ++d) s12 += xs[clampi(t - 5 + d, 0, Lsz - 1) * 33 + j];
      float trend = (s4 * 0.25f + s8 * 0.125f + s12 * (1.0f / 12.0f)) * (1.0f / 3.0f);
      float sea = 0.0f;
      for (int c = 0; c < 3; ++c) {
        int k = selK[j * 3 + c];
        int idx = (k * t) % Lsz;
        sea += selRe[j * 3 + c] * cosT[idx] - selIm[j * 3 + c] * sinT[idx];
      }
      sea *= (2.0f / (float)Lsz);
      float val = xs[t * 33 + j] + trend + sea;
      acc = fmaf(val, w_start[n], acc);
    }
    gxpart[((size_t)chunk * Bsz + b) * Lsz + t] = acc;
  }
}

// ---------------------------------------------------------------------------
// Kernel 2: reduce gx partials, gating logits, top-2 softmax gates,
// balance loss (cv^2 of importance + load).
// ---------------------------------------------------------------------------
__global__ __launch_bounds__(256)
void k_gate(const float* __restrict__ gxpart, const float* __restrict__ b_start,
            const float* __restrict__ w_gate, float* __restrict__ gates,
            float* __restrict__ loss_out) {
  __shared__ float gxs[Bsz * Lsz];      // 48 KB
  __shared__ float logits[Bsz * Esz];
  __shared__ float g[Bsz * Esz];
  const int tid = threadIdx.x;

  for (int i = tid; i < Bsz * Lsz; i += 256) {
    float s = 0.0f;
    for (int c = 0; c < NCHUNK; ++c) s += gxpart[(size_t)c * Bsz * Lsz + i];
    gxs[i] = s;
  }
  __syncthreads();

  {
    const float bs = b_start[0];
    const int b = tid >> 2, e = tid & 3;
    float acc = 0.0f;
    for (int l = 0; l < Lsz; ++l)
      acc = fmaf(gxs[b * Lsz + l] + bs, w_gate[l * Esz + e], acc);
    logits[tid] = acc;
  }
  __syncthreads();

  if (tid < Bsz) {
    const int b = tid;
    float v[4];
    for (int e = 0; e < 4; ++e) v[e] = logits[b * 4 + e];
    int i0 = 0;
    for (int e = 1; e < 4; ++e) if (v[e] > v[i0]) i0 = e;
    int i1 = -1; float best = -3.4e38f;
    for (int e = 0; e < 4; ++e) if (e != i0 && v[e] > best) { best = v[e]; i1 = e; }
    float p1 = __expf(v[i1] - v[i0]);     // softmax over top-2, max-subtracted
    float s = 1.0f + p1;
    for (int e = 0; e < 4; ++e) g[b * 4 + e] = 0.0f;
    g[b * 4 + i0] = 1.0f / s;
    g[b * 4 + i1] = p1 / s;
  }
  __syncthreads();
  if (tid < Bsz * Esz) gates[tid] = g[tid];

  if (tid == 0) {
    float loss = 0.0f;
    for (int pass = 0; pass < 2; ++pass) {
      float v[4];
      for (int e = 0; e < 4; ++e) {
        float s = 0.0f;
        for (int b = 0; b < Bsz; ++b) {
          float gv = g[b * 4 + e];
          s += (pass == 0) ? gv : ((gv > 0.0f) ? 1.0f : 0.0f);
        }
        v[e] = s;
      }
      float mean = (v[0] + v[1] + v[2] + v[3]) * 0.25f;
      float var = 0.0f;
      for (int e = 0; e < 4; ++e) { float d = v[e] - mean; var += d * d; }
      var *= (1.0f / 3.0f);                       // ddof=1, E=4
      loss += var / (mean * mean + 1e-10f);
    }
    loss_out[0] = 0.01f * loss;
  }
}

// ---------------------------------------------------------------------------
// Kernel 3: Weff[b][l][m] = sum_e g*W_exp[e][l][m] + W_end[m][l]  (note 'ol'!)
//           beff[b][m]    = sum_e g*b_exp[e][m]    + b_end[m]
// ---------------------------------------------------------------------------
__global__ __launch_bounds__(256)
void k_weff(const float* __restrict__ gates, const float* __restrict__ W_exp,
            const float* __restrict__ b_exp, const float* __restrict__ W_end,
            const float* __restrict__ b_end, float* __restrict__ Weff,
            float* __restrict__ beff) {
  const int b = blockIdx.x;
  const int tid = threadIdx.x;
  const float g0 = gates[b * 4 + 0], g1 = gates[b * 4 + 1];
  const float g2 = gates[b * 4 + 2], g3 = gates[b * 4 + 3];
  for (int i = tid; i < Lsz * Lsz; i += 256) {
    int l = i / Lsz, m = i - l * Lsz;
    float w = W_end[m * Lsz + l];
    w = fmaf(g0, W_exp[0 * Lsz * Lsz + i], w);
    w = fmaf(g1, W_exp[1 * Lsz * Lsz + i], w);
    w = fmaf(g2, W_exp[2 * Lsz * Lsz + i], w);
    w = fmaf(g3, W_exp[3 * Lsz * Lsz + i], w);
    Weff[(size_t)b * Lsz * Lsz + i] = w;
  }
  for (int m = tid; m < Lsz; m += 256) {
    float v = b_end[m];
    v = fmaf(g0, b_exp[0 * Lsz + m], v);
    v = fmaf(g1, b_exp[1 * Lsz + m], v);
    v = fmaf(g2, b_exp[2 * Lsz + m], v);
    v = fmaf(g3, b_exp[3 * Lsz + m], v);
    beff[b * Lsz + m] = v;
  }
}

// ---------------------------------------------------------------------------
// Kernel 4: out[b,m,n] = sum_l Weff[b][l][m] * x[b,l,n] + beff[b][m]
// WMMA f32 = f16 x f16: each 256-thread block = 16(m) x 128(n) tile,
// 8 waves x one 16x16 WMMA tile, K=192 in 6 steps of 32.
// ---------------------------------------------------------------------------
__global__ __launch_bounds__(256)
void k_moe_gemm(const float* __restrict__ x, const float* __restrict__ Weff,
                const float* __restrict__ beff, float* __restrict__ out) {
  const int b  = blockIdx.z;
  const int m0 = blockIdx.y * 16;
  const int n0 = blockIdx.x * 128;
  const int tid = threadIdx.x;

  __shared__ _Float16 aT[16 * LP];     // A = Weff^T tile: [m][l], l contiguous
  __shared__ _Float16 bT[128 * LP];    // B = x tile:      [n][l], l contiguous

  __builtin_prefetch(&Weff[(size_t)b * Lsz * Lsz + (tid & 191) * Lsz + m0], 0, 1);

  // A tile: f32 -> f16 at LDS store time
  for (int e = tid; e < Lsz * 16; e += 256) {
    int l = e >> 4, m = e & 15;
    float w = Weff[(size_t)b * Lsz * Lsz + l * Lsz + (m0 + m)];
    aT[m * LP + l] = (_Float16)w;
  }
  // B tile: 128 consecutive n per l row -> fully coalesced; zero-pad n >= N
  for (int e = tid; e < Lsz * 128; e += 256) {
    int l = e >> 7, j = e & 127;
    int n = n0 + j;
    float v = (n < Nsz) ? x[(size_t)(b * Lsz + l) * Nsz + n] : 0.0f;
    bT[j * LP + l] = (_Float16)v;
  }
  __syncthreads();

  const int lane = tid & 31;
  const int wv   = tid >> 5;
  const int half = lane >> 4;        // lane group (ISA 16-bit A/B layouts)
  const int mr   = lane & 15;        // A row
  const int nl   = lane & 15;        // B/D column
  const _Float16* aBase = &aT[mr * LP];
  const _Float16* bBase = &bT[(wv * 16 + nl) * LP];

  v8f acc;
  #pragma unroll
  for (int r = 0; r < 8; ++r)        // C init = bias (D row = r + 8*half)
    acc[r] = beff[b * Lsz + m0 + r + 8 * half];

  #pragma unroll
  for (int kk = 0; kk < Lsz / 32; ++kk) {
    const int k0 = kk * 32;
    AV a, bb;
    // A 16-bit 16x32: lanes0-15 VGPR v: K = 2v,2v+1 (v<4), 16+2(v-4) (v>=4);
    // lanes16-31: +8.  Pairs are contiguous in [m][l] layout -> b32/b128 reads.
    #pragma unroll
    for (int vI = 0; vI < 4; ++vI) {
      a.u[vI]     = *(const unsigned int*)(aBase + k0 + half * 8 + 2 * vI);
      a.u[4 + vI] = *(const unsigned int*)(aBase + k0 + 16 + half * 8 + 2 * vI);
    }
    // B 16-bit 32x16: same K order as A: VGPR v: K = 2v,2v+1 (v<4), 16+2(v-4) (v>=4); lanes16-31: +8
    #pragma unroll
    for (int vI = 0; vI < 4; ++vI) {
      bb.u[vI]     = *(const unsigned int*)(bBase + k0 + half * 8 + 2 * vI);
      bb.u[4 + vI] = *(const unsigned int*)(bBase + k0 + 16 + half * 8 + 2 * vI);
    }

    acc = __builtin_amdgcn_wmma_f32_16x16x32_f16(
        false, a.v, false, bb.v, (short)0, acc, false, false);
  }

  const int n = n0 + wv * 16 + nl;
  if (n < Nsz) {
    #pragma unroll
    for (int r = 0; r < 8; ++r) {
      int m = m0 + r + 8 * half;
      out[(size_t)(b * Lsz + m) * Nsz + n] = acc[r];
    }
  }
}

// ---------------------------------------------------------------------------
extern "C" void kernel_launch(void* const* d_in, const int* in_sizes, int n_in,
                              void* d_out, int out_size, void* d_ws, size_t ws_size,
                              hipStream_t stream) {
  (void)in_sizes; (void)n_in; (void)out_size; (void)ws_size;

  const float* x       = (const float*)d_in[0];
  const float* w_start = (const float*)d_in[1];
  const float* b_start = (const float*)d_in[2];
  const float* w_gate  = (const float*)d_in[3];
  const float* W_exp   = (const float*)d_in[4];
  const float* b_exp   = (const float*)d_in[5];
  const float* W_end   = (const float*)d_in[6];
  const float* b_end   = (const float*)d_in[7];

  float* out = (float*)d_out;
  float* loss_out = out + (size_t)Bsz * Lsz * Nsz;   // scalar after the tensor

  float* ws = (float*)d_ws;
  float* gxpart = ws;                                  // 27*64*192
  float* gates  = gxpart + (size_t)NCHUNK * Bsz * Lsz; // 256
  float* beff   = gates + Bsz * Esz;                   // 64*192
  float* Weff   = beff + Bsz * Lsz;                    // 64*192*192

  k_decomp<<<dim3(NCHUNK, Bsz), 256, 0, stream>>>(x, w_start, gxpart);
  k_gate<<<1, 256, 0, stream>>>(gxpart, b_start, w_gate, gates, loss_out);
  k_weff<<<Bsz, 256, 0, stream>>>(gates, W_exp, b_exp, W_end, b_end, Weff, beff);
  k_moe_gemm<<<dim3((Nsz + 127) / 128, Lsz / 16, Bsz), 256, 0, stream>>>(
      x, Weff, beff, out);
}
